// SequenceMemoryUpdater_75265006895946
// MI455X (gfx1250) — matrix operand order repaired and checked by reading, not compiled, not run on hardware
//
#include <hip/hip_runtime.h>

typedef __attribute__((ext_vector_type(16))) __bf16 v16bf;
typedef __attribute__((ext_vector_type(8)))  float  v8f;

#define NU        262144
#define MSG_DIM   512
#define MEM_DIM   256
#define BLOCK_M   64
#define NJT       48            // 768/16 gate-column tiles
#define NKS       24            // 512/32 (x) + 256/32 (h) K-steps
#define NWAVE     16
#define THREADS   (NWAVE*32)
#define LSTRIDE   12            // u32 per lane block in LDS (8 data + 4 pad)
#define AMS       (NKS*32*LSTRIDE)   // u32 per M-subtile plane  = 9216
#define AKS       (32*LSTRIDE)       // u32 per K-step           = 384
#define APLANE    (4*AMS)            // u32 per plane            = 36864
#define BKS       256                // u32 per K-step per gate tile in ws
#define BJT       (NKS*BKS)          // u32 per gate tile        = 6144

__device__ __forceinline__ unsigned f2bf(float f) {
  unsigned u = __float_as_uint(f);
  u += 0x7FFFu + ((u >> 16) & 1u);          // round-to-nearest-even
  return u >> 16;
}
__device__ __forceinline__ float bfbits2f(unsigned hs) {
  return __uint_as_float(hs << 16);
}

union Frag { v16bf v; unsigned w[8]; uint4 q[2]; };

__device__ __forceinline__ v8f wmma_acc(const Frag& a, const Frag& b, v8f c) {
  return __builtin_amdgcn_wmma_f32_16x16x32_bf16(false, a.v, false, b.v,
                                                 (short)0, c, false, false);
}

// ---------------------------------------------------------------- copy bank
__global__ void copy_bank(const float4* __restrict__ src,
                          float4* __restrict__ dst, long long n4) {
  long long i = (long long)blockIdx.x * blockDim.x + threadIdx.x;
  long long stride = (long long)gridDim.x * blockDim.x;
  for (; i < n4; i += stride) dst[i] = src[i];
}

// ------------------------------------------- prepack W into B-fragment layout
// t = (jt*NKS + ks)*32 + lane.  B element e (0..15) of lane (col=lane&15,
// half=lane>>4) holds W[jt*16+col][ks*32 + 16*half + e].  ks 0..15 -> W_ih,
// ks 16..23 -> W_hh.
__global__ void prepack_weights(const float* __restrict__ Wih,
                                const float* __restrict__ Whh,
                                unsigned* __restrict__ wsH,
                                unsigned* __restrict__ wsL) {
  int t = blockIdx.x * blockDim.x + threadIdx.x;
  if (t >= NJT * NKS * 32) return;
  int lane = t & 31;
  int ks   = (t >> 5) % NKS;
  int jt   = t / (NKS * 32);
  int j    = jt * 16 + (lane & 15);
  int half = lane >> 4;
  const float* src = (ks < 16)
      ? (Wih + (size_t)j * MSG_DIM + ks * 32 + 16 * half)
      : (Whh + (size_t)j * MEM_DIM + (ks - 16) * 32 + 16 * half);
  unsigned* oh = wsH + (size_t)t * 8;
  unsigned* ol = wsL + (size_t)t * 8;
#pragma unroll
  for (int p = 0; p < 8; ++p) {
    float f0 = src[2 * p], f1 = src[2 * p + 1];
    unsigned h0 = f2bf(f0), h1 = f2bf(f1);
    unsigned l0 = f2bf(f0 - bfbits2f(h0));
    unsigned l1 = f2bf(f1 - bfbits2f(h1));
    oh[p] = h0 | (h1 << 16);
    ol[p] = l0 | (l1 << 16);
  }
}

// stage one float4 (4 consecutive K elems) of A into frag-layout LDS planes
__device__ __forceinline__ void stage4(unsigned* smem, int row, int ks,
                                       int kin /*0..31, %4==0*/, float4 v) {
  int half  = kin >> 4;
  int e0    = kin & 15;                          // element 0..15, %4==0
  int lane  = (row & 15) + (half << 4);
  int ms    = row >> 4;
  unsigned base = (unsigned)ms * AMS + (unsigned)ks * AKS
                + (unsigned)lane * LSTRIDE + (unsigned)(e0 >> 1);
  unsigned h0 = f2bf(v.x), h1 = f2bf(v.y), h2 = f2bf(v.z), h3 = f2bf(v.w);
  unsigned l0 = f2bf(v.x - bfbits2f(h0)), l1 = f2bf(v.y - bfbits2f(h1));
  unsigned l2 = f2bf(v.z - bfbits2f(h2)), l3 = f2bf(v.w - bfbits2f(h3));
  uint2 ph; ph.x = h0 | (h1 << 16); ph.y = h2 | (h3 << 16);
  uint2 pl; pl.x = l0 | (l1 << 16); pl.y = l2 | (l3 << 16);
  *(uint2*)(smem + base)          = ph;
  *(uint2*)(smem + APLANE + base) = pl;
}

__device__ __forceinline__ void load_afrag(const unsigned* smem, unsigned idx,
                                           Frag& ah, Frag& al) {
  ah.q[0] = *(const uint4*)(smem + idx);
  ah.q[1] = *(const uint4*)(smem + idx + 4);
  al.q[0] = *(const uint4*)(smem + APLANE + idx);
  al.q[1] = *(const uint4*)(smem + APLANE + idx + 4);
}

__device__ __forceinline__ void load_bfrag(const unsigned* p, Frag& f) {
  f.q[0] = *(const uint4*)(p);
  f.q[1] = *(const uint4*)(p + 4);
}

// ------------------------------------------------------------- fused GRU GEMM
__global__ void __launch_bounds__(THREADS)
gru_update(const long long* __restrict__ ids,
           const float* __restrict__ msgs,
           const float* __restrict__ bank,
           const float* __restrict__ b_ih,
           const float* __restrict__ b_hh,
           const unsigned* __restrict__ wsH,
           const unsigned* __restrict__ wsL,
           float* __restrict__ out) {
  extern __shared__ unsigned smem[];   // A fragments: hi plane | lo plane
  const int tid  = threadIdx.x;
  const int row0 = blockIdx.x * BLOCK_M;

  // ---- stage x (messages): K-steps 0..15
  for (int it = tid; it < BLOCK_M * (MSG_DIM / 4); it += THREADS) {
    int r = it >> 7, c4 = it & 127;
    float4 v = *(const float4*)(msgs + (size_t)(row0 + r) * MSG_DIM + c4 * 4);
    stage4(smem, r, c4 >> 3, (c4 & 7) * 4, v);
  }
  // ---- stage gathered h rows: K-steps 16..23
  for (int it = tid; it < BLOCK_M * (MEM_DIM / 4); it += THREADS) {
    int r = it >> 6, c4 = it & 63;
    long long id = ids[row0 + r];
    float4 v = *(const float4*)(bank + (size_t)id * MEM_DIM + c4 * 4);
    stage4(smem, r, 16 + (c4 >> 3), (c4 & 7) * 4, v);
  }
  __syncthreads();

  const int lane = tid & 31;
  const int wv   = tid >> 5;        // this wave's d_tile (0..15)
  const int col  = lane & 15;
  const int half = lane >> 4;
  const int d    = wv * 16 + col;   // output feature index 0..255

  const float bir = b_ih[d],        bhr = b_hh[d];
  const float biz = b_ih[256 + d],  bhz = b_hh[256 + d];
  const float bin_ = b_ih[512 + d], bhn = b_hh[512 + d];

  const unsigned laneB = (unsigned)lane * 8;
  const unsigned* bHr = wsH + (unsigned)(wv)      * BJT + laneB;
  const unsigned* bHz = wsH + (unsigned)(16 + wv) * BJT + laneB;
  const unsigned* bHn = wsH + (unsigned)(32 + wv) * BJT + laneB;
  const unsigned* bLr = wsL + (unsigned)(wv)      * BJT + laneB;
  const unsigned* bLz = wsL + (unsigned)(16 + wv) * BJT + laneB;
  const unsigned* bLn = wsL + (unsigned)(32 + wv) * BJT + laneB;
  const unsigned laneA = (unsigned)lane * LSTRIDE;

#pragma unroll
  for (int b = 0; b < 2; ++b) {
    v8f ar[2], az[2], an[2], ahn[2];
#pragma unroll
    for (int m = 0; m < 2; ++m) { ar[m] = {}; az[m] = {}; an[m] = {}; ahn[m] = {}; }
    const unsigned aMs = (unsigned)(b * 2) * AMS + laneA;

    // ---- pass X: K-steps 0..15 -> accumulate r, z, i_n
    for (int ks = 0; ks < 16; ++ks) {
      const unsigned bo = (unsigned)ks * BKS;
      Frag brh, brl, bzh, bzl, bnh, bnl;
      load_bfrag(bHr + bo, brh); load_bfrag(bLr + bo, brl);
      load_bfrag(bHz + bo, bzh); load_bfrag(bLz + bo, bzl);
      load_bfrag(bHn + bo, bnh); load_bfrag(bLn + bo, bnl);
      __builtin_prefetch(bHr + bo + BKS, 0, 1);
      __builtin_prefetch(bLr + bo + BKS, 0, 1);
#pragma unroll
      for (int m = 0; m < 2; ++m) {
        Frag ah, al;
        load_afrag(smem, aMs + (unsigned)m * AMS + (unsigned)ks * AKS, ah, al);
        ar[m] = wmma_acc(ah, brh, ar[m]);
        ar[m] = wmma_acc(ah, brl, ar[m]);
        ar[m] = wmma_acc(al, brh, ar[m]);
        az[m] = wmma_acc(ah, bzh, az[m]);
        az[m] = wmma_acc(ah, bzl, az[m]);
        az[m] = wmma_acc(al, bzh, az[m]);
        an[m] = wmma_acc(ah, bnh, an[m]);
        an[m] = wmma_acc(ah, bnl, an[m]);
        an[m] = wmma_acc(al, bnh, an[m]);
      }
    }
    // ---- pass H: K-steps 16..23 -> accumulate r, z, h_n
    for (int ks = 16; ks < 24; ++ks) {
      const unsigned bo = (unsigned)ks * BKS;
      Frag brh, brl, bzh, bzl, bnh, bnl;
      load_bfrag(bHr + bo, brh); load_bfrag(bLr + bo, brl);
      load_bfrag(bHz + bo, bzh); load_bfrag(bLz + bo, bzl);
      load_bfrag(bHn + bo, bnh); load_bfrag(bLn + bo, bnl);
#pragma unroll
      for (int m = 0; m < 2; ++m) {
        Frag ah, al;
        load_afrag(smem, aMs + (unsigned)m * AMS + (unsigned)ks * AKS, ah, al);
        ar[m]  = wmma_acc(ah, brh, ar[m]);
        ar[m]  = wmma_acc(ah, brl, ar[m]);
        ar[m]  = wmma_acc(al, brh, ar[m]);
        az[m]  = wmma_acc(ah, bzh, az[m]);
        az[m]  = wmma_acc(ah, bzl, az[m]);
        az[m]  = wmma_acc(al, bzh, az[m]);
        ahn[m] = wmma_acc(ah, bnh, ahn[m]);
        ahn[m] = wmma_acc(ah, bnl, ahn[m]);
        ahn[m] = wmma_acc(al, bnh, ahn[m]);
      }
    }

    // ---- gates + scatter.  C/D: element i -> row i + 8*half, col = lane&15.
#pragma unroll
    for (int m = 0; m < 2; ++m) {
#pragma unroll
      for (int i = 0; i < 8; ++i) {
        int rloc = (b * 2 + m) * 16 + i + 8 * half;
        long long id = ids[row0 + rloc];
        float h  = bank[(size_t)id * MEM_DIM + d];
        float r  = 1.f / (1.f + __expf(-(ar[m][i] + bir + bhr)));
        float z  = 1.f / (1.f + __expf(-(az[m][i] + biz + bhz)));
        float nn = tanhf(an[m][i] + bin_ + r * (ahn[m][i] + bhn));
        out[(size_t)id * MEM_DIM + d] = (1.f - z) * nn + z * h;
      }
    }
  }
}

// --------------------------------------------------------------------- launch
extern "C" void kernel_launch(void* const* d_in, const int* in_sizes, int n_in,
                              void* d_out, int out_size, void* d_ws, size_t ws_size,
                              hipStream_t stream) {
  const long long* ids  = (const long long*)d_in[0];
  const float*     msgs = (const float*)d_in[1];
  const float*     bank = (const float*)d_in[2];
  const float*     Wih  = (const float*)d_in[3];
  const float*     Whh  = (const float*)d_in[4];
  const float*     bih  = (const float*)d_in[5];
  const float*     bhh  = (const float*)d_in[6];
  float* out = (float*)d_out;

  unsigned* wsH = (unsigned*)d_ws;                       // 1.18 MB hi frags
  unsigned* wsL = wsH + (size_t)NJT * NKS * 32 * 8;      // 1.18 MB lo frags

  long long n4 = (long long)out_size / 4;
  copy_bank<<<8192, 256, 0, stream>>>((const float4*)bank, (float4*)d_out, n4);
  prepack_weights<<<(NJT * NKS * 32 + 255) / 256, 256, 0, stream>>>(Wih, Whh, wsH, wsL);

  int ldsB = 2 * APLANE * 4;                             // 294912 B (<320 KB)
  (void)hipFuncSetAttribute((const void*)gru_update,
                            hipFuncAttributeMaxDynamicSharedMemorySize, ldsB);
  gru_update<<<NU / BLOCK_M, THREADS, ldsB, stream>>>(ids, msgs, bank, bih, bhh,
                                                      wsH, wsL, out);
}
